// MambaEncoder_71648644432337
// MI455X (gfx1250) — compile-verified, matrix operand order checked by reading, not run on hardware
//
#include <hip/hip_runtime.h>

// ---------------------------------------------------------------------------
// CDNA5 (gfx1250) Mamba encoder. GEMMs via v_wmma_f32_16x16x32_bf16 (wave32),
// W tiles staged in LDS with double-buffered async Global->LDS DMA.
// ---------------------------------------------------------------------------

typedef __attribute__((ext_vector_type(16))) __bf16 v16bf;
typedef __attribute__((ext_vector_type(8)))  float  v8f;

__device__ __forceinline__ unsigned short f2b(float f) {
  unsigned int u = __float_as_uint(f);
  u += 0x7FFFu + ((u >> 16) & 1u);   // round-to-nearest-even
  return (unsigned short)(u >> 16);
}

union FragBF { v16bf v; unsigned int u[8]; };

#if __has_builtin(__builtin_amdgcn_global_load_async_to_lds_b128)
#define ASYNC_LDS 1
#else
#define ASYNC_LDS 0
#endif

typedef __attribute__((__vector_size__(16))) int v4i_;
typedef __attribute__((address_space(1))) v4i_ gv4i;          // global int4
typedef __attribute__((address_space(3))) v4i_ lv4i;          // LDS int4

// copy 16 bytes global -> LDS (async DMA when available)
__device__ __forceinline__ void fill16(const unsigned short* g, unsigned short* l) {
#if ASYNC_LDS
  __builtin_amdgcn_global_load_async_to_lds_b128(
      (gv4i*)(const gv4i*)g,   // AS1 src (const peeled by C-style cast)
      (lv4i*)l,                // AS3 dst
      0, 0);
#else
  *(uint4*)l = *(const uint4*)g;
#endif
}

__device__ __forceinline__ void async_wait_all() {
#if ASYNC_LDS
#if __has_builtin(__builtin_amdgcn_s_wait_asynccnt)
  __builtin_amdgcn_s_wait_asynccnt(0);
#else
  asm volatile("s_wait_asynccnt 0x0" ::: "memory");
#endif
#endif
}

#define WROW 40   // LDS row stride in ushorts (32 data + 8 pad: bank-conflict free)

// C[M,N] = act(A[M,K]_bf16 * W[N,K]_bf16^T + bias)
// mode: 0 = none, 1 = bias + softplus, 2 = bias only
// Grid: ((N+63)/64, (M+63)/64), block 128 (4 waves). M multiple of 64, K of 32.
__global__ __launch_bounds__(128)
void k_gemm(const unsigned short* __restrict__ A,
            const unsigned short* __restrict__ W,
            const float* __restrict__ bias,
            float* __restrict__ C,
            int M, int N, int K, int mode)
{
  __shared__ __align__(16) unsigned short sW[2][64 * WROW];
  const int tid  = threadIdx.x;
  const int lane = tid & 31;
  const int wave = tid >> 5;
  const int g    = lane >> 4;      // lane group selects K offset +8
  const int lr   = lane & 15;
  const int m0   = blockIdx.y * 64 + wave * 16;
  const int n0   = blockIdx.x * 64;

  // cooperative W-tile fill: 64 rows x 32 k (4KB data), 2 x 16B chunks / thread
  auto fill = [&](int buf, int k0) {
#pragma unroll
    for (int j = 0; j < 2; ++j) {
      const int chunk = tid * 2 + j;        // 0..255
      const int row = chunk >> 2;
      const int q   = chunk & 3;
      int n = n0 + row; if (n >= N) n = N - 1;
      fill16(W + (size_t)n * K + k0 + q * 8, &sW[buf][row * WROW + q * 8]);
    }
  };

  v8f acc[4] = {};
  const unsigned short* arow = A + (size_t)(m0 + lr) * K + 8 * g;

  fill(0, 0);
  int buf = 0;
  for (int k0 = 0; k0 < K; k0 += 32) {
    // A fragment (per ISA 16-bit A layout): two 16B chunks per lane
    FragBF a;
#pragma unroll
    for (int v = 0; v < 8; ++v) {
      const int koff = (v < 4) ? (2 * v) : (16 + 2 * (v - 4));
      a.u[v] = *(const unsigned int*)(arow + k0 + koff);
    }
    if (k0 + 64 < K) __builtin_prefetch(arow + k0 + 64, 0, 0);

    async_wait_all();
    __syncthreads();                        // tile(buf) visible to all waves
    if (k0 + 32 < K) fill(buf ^ 1, k0 + 32);

    // gather all 4 B fragments first (one DS-wait), then 4 back-to-back WMMAs
    FragBF b[4];
#pragma unroll
    for (int ns = 0; ns < 4; ++ns) {
      const int row32 = (ns * 16 + lr) * WROW;
      uint4 lo = *(const uint4*)&sW[buf][row32 + 8 * g];
      uint4 hi = *(const uint4*)&sW[buf][row32 + 16 + 8 * g];
      b[ns].u[0] = lo.x; b[ns].u[1] = lo.y; b[ns].u[2] = lo.z; b[ns].u[3] = lo.w;
      b[ns].u[4] = hi.x; b[ns].u[5] = hi.y; b[ns].u[6] = hi.z; b[ns].u[7] = hi.w;
    }
#pragma unroll
    for (int ns = 0; ns < 4; ++ns) {
      acc[ns] = __builtin_amdgcn_wmma_f32_16x16x32_bf16(
          false, a.v, false, b[ns].v, (short)0, acc[ns], false, false);
    }
    __syncthreads();                        // all reads done before refill
    buf ^= 1;
  }

  // D layout: vgpr r, lane L -> (M = r + 8*(L>>4), N = L&15)
#pragma unroll
  for (int ns = 0; ns < 4; ++ns) {
#pragma unroll
    for (int r = 0; r < 8; ++r) {
      const int m = m0 + r + 8 * g;
      const int n = n0 + ns * 16 + lr;
      if (m < M && n < N) {
        float v = acc[ns][r];
        if (mode) {
          v += bias[n];
          if (mode == 1) v = (v > 20.f) ? v : __logf(1.f + __expf(v));
        }
        C[(size_t)m * N + n] = v;
      }
    }
  }
}

// f32 -> bf16 with row stride + K zero-padding (weights, activations, dt split)
__global__ void k_cvt(const float* __restrict__ in, unsigned short* __restrict__ out,
                      int rows, int inStride, int K, int Kp)
{
  int i = blockIdx.x * blockDim.x + threadIdx.x;
  if (i >= rows * Kp) return;
  int k = i % Kp, r = i / Kp;
  float v = (k < K) ? in[(size_t)r * inStride + k] : 0.f;
  out[i] = f2b(v);
}

// causal depthwise conv (dc=4) + bias + SiLU; reads xc half of xz [*,1024]
__global__ void k_conv(const float* __restrict__ xz, const float* __restrict__ cw,
                       const float* __restrict__ cb, float* __restrict__ xcf,
                       unsigned short* __restrict__ xcb, int batch, int L)
{
  int i = blockIdx.x * blockDim.x + threadIdx.x;
  if (i >= batch * L * 512) return;
  int c = i & 511; int t = i >> 9; int l = t % L; int b = t / L;
  float s = cb[c];
#pragma unroll
  for (int j = 0; j < 4; ++j) {
    int ll = l - 3 + j;
    if (ll >= 0) s += xz[((size_t)(b * L + ll)) * 1024 + c] * cw[c * 4 + j];
  }
  s = s / (1.f + __expf(-s));
  size_t row = (size_t)(b * L + l);
  xcf[row * 512 + c] = s;
  xcb[row * 512 + c] = f2b(s);
}

// selective scan fused with SiLU(z) gating; writes bf16 y for out_proj GEMM
__global__ void k_scan(const float* __restrict__ xc, const float* __restrict__ xdbl,
                       const float* __restrict__ dtb, const float* __restrict__ xz,
                       const float* __restrict__ Alog, const float* __restrict__ Dp,
                       unsigned short* __restrict__ yb, int batch, int L)
{
  int i = blockIdx.x * blockDim.x + threadIdx.x;
  if (i >= batch * 512) return;
  int d = i & 511; int b = i >> 9;
  float A[16], h[16];
#pragma unroll
  for (int s = 0; s < 16; ++s) { A[s] = -__expf(Alog[d * 16 + s]); h[s] = 0.f; }
  float Dd = Dp[d];
  for (int l = 0; l < L; ++l) {
    size_t row = (size_t)b * L + l;
    float dt = dtb[row * 512 + d];
    float u  = xc[row * 512 + d];
    const float* bc = xdbl + row * 48;
    float du = dt * u;
    float y = 0.f;
#pragma unroll
    for (int s = 0; s < 16; ++s) {
      h[s] = __expf(dt * A[s]) * h[s] + du * bc[16 + s];
      y += h[s] * bc[32 + s];
    }
    y += u * Dd;
    float z = xz[row * 1024 + 512 + d];
    float gt = z / (1.f + __expf(-z));
    yb[row * 512 + d] = f2b(y * gt);
  }
}

// segment-merge LayerNorm over 512, writes normalized bf16 rows
__global__ __launch_bounds__(128)
void k_merge_norm(const float* __restrict__ x, const float* __restrict__ g,
                  const float* __restrict__ bb, unsigned short* __restrict__ o,
                  int rows)
{
  __shared__ float s1[128], s2[128];
  int row = blockIdx.x; int tid = threadIdx.x;
  if (row >= rows) return;
  const float* r = x + (size_t)row * 512;
  float a = 0.f, q = 0.f;
  for (int i = tid; i < 512; i += 128) { float v = r[i]; a += v; q += v * v; }
  s1[tid] = a; s2[tid] = q; __syncthreads();
  for (int st = 64; st > 0; st >>= 1) {
    if (tid < st) { s1[tid] += s1[tid + st]; s2[tid] += s2[tid + st]; }
    __syncthreads();
  }
  float mu = s1[0] * (1.f / 512.f);
  float var = s2[0] * (1.f / 512.f) - mu * mu;
  float rs = rsqrtf(var + 1e-5f);
  for (int i = tid; i < 512; i += 128)
    o[(size_t)row * 512 + i] = f2b((r[i] - mu) * rs * g[i] + bb[i]);
}

// [B,D,sn,dm] f32 -> [B,sn,D,dm] bf16 (dim-stage input)
__global__ void k_tr2b(const float* __restrict__ x, unsigned short* __restrict__ o,
                       int Bb, int D, int sn, int dm)
{
  int i = blockIdx.x * blockDim.x + threadIdx.x;
  if (i >= Bb * D * sn * dm) return;
  int c = i % dm; int t = i / dm; int d = t % D; t /= D; int l = t % sn; int b = t / sn;
  o[i] = f2b(x[(((size_t)b * D + d) * sn + l) * dm + c]);
}

// [B,sn,D,dm] f32 -> [B,D,sn,dm] f32 (dim-stage output)
__global__ void k_trback(const float* __restrict__ in, float* __restrict__ out,
                         int Bb, int D, int sn, int dm)
{
  int i = blockIdx.x * blockDim.x + threadIdx.x;
  if (i >= Bb * D * sn * dm) return;
  int c = i % dm; int t = i / dm; int l = t % sn; t /= sn; int d = t % D; int b = t / D;
  out[i] = in[(((size_t)b * sn + l) * D + d) * dm + c];
}

// ---------------------------------------------------------------------------

extern "C" void kernel_launch(void* const* d_in, const int* in_sizes, int n_in,
                              void* d_out, int out_size, void* d_ws, size_t ws_size,
                              hipStream_t stream)
{
  (void)in_sizes; (void)n_in; (void)out_size; (void)ws_size;
  const int Bb = 8, NV = 16, SEG0 = 128, DMD = 256;
  const int Mmax = Bb * NV * SEG0;  // 16384

  char* base = (char*)d_ws;
  size_t off = 0;
  auto alloc = [&](size_t bytes) -> void* {
    void* p = base + off; off = (off + bytes + 255) & ~(size_t)255; return p;
  };
  float* f_x    = (float*)alloc((size_t)Mmax * 256 * 4);
  float* f_xz   = (float*)alloc((size_t)Mmax * 1024 * 4);
  float* f_xc   = (float*)alloc((size_t)Mmax * 512 * 4);
  float* f_xdbl = (float*)alloc((size_t)Mmax * 48 * 4);
  float* f_dt   = (float*)alloc((size_t)Mmax * 512 * 4);
  float* f_o    = (float*)alloc((size_t)Mmax * 256 * 4);
  unsigned short* b_x   = (unsigned short*)alloc((size_t)Mmax * 256 * 2);
  unsigned short* b_xc  = (unsigned short*)alloc((size_t)Mmax * 512 * 2);
  unsigned short* b_dt  = (unsigned short*)alloc((size_t)Mmax * 32 * 2);
  unsigned short* b_y   = (unsigned short*)alloc((size_t)Mmax * 512 * 2);
  unsigned short* b_win = (unsigned short*)alloc((size_t)1024 * 256 * 2);
  unsigned short* b_wxp = (unsigned short*)alloc((size_t)48 * 512 * 2);
  unsigned short* b_wdt = (unsigned short*)alloc((size_t)512 * 32 * 2);
  unsigned short* b_wout= (unsigned short*)alloc((size_t)256 * 512 * 2);
  unsigned short* b_wlin= (unsigned short*)alloc((size_t)256 * 512 * 2);

  auto cvt = [&](const float* in, unsigned short* out, int rows, int stride, int K, int Kp) {
    int total = rows * Kp;
    k_cvt<<<dim3((total + 255) / 256), dim3(256), 0, stream>>>(in, out, rows, stride, K, Kp);
  };
  auto gemm = [&](const unsigned short* A, const unsigned short* W, const float* bias,
                  float* C, int M, int N, int K, int mode) {
    k_gemm<<<dim3((N + 63) / 64, (M + 63) / 64), dim3(128), 0, stream>>>(A, W, bias, C, M, N, K, mode);
  };
  // p[0..8] = in_proj, conv_w, conv_b, x_proj, dt_w, dt_b, A_log, D, out_proj
  auto mamba = [&](const float* const* p, const unsigned short* bx_in,
                   int batch, int L, float* outbuf) {
    int M = batch * L;
    cvt(p[0], b_win, 1024, 256, 256, 256);
    gemm(bx_in, b_win, nullptr, f_xz, M, 1024, 256, 0);
    int tot = M * 512;
    k_conv<<<dim3((tot + 255) / 256), dim3(256), 0, stream>>>(f_xz, p[1], p[2], f_xc, b_xc, batch, L);
    cvt(p[3], b_wxp, 48, 512, 512, 512);
    gemm(b_xc, b_wxp, nullptr, f_xdbl, M, 48, 512, 0);
    cvt(f_xdbl, b_dt, M, 48, 16, 32);          // dt slice, K padded 16->32
    cvt(p[4], b_wdt, 512, 16, 16, 32);
    gemm(b_dt, b_wdt, p[5], f_dt, M, 512, 32, 1);   // softplus epilogue
    int ns = batch * 512;
    k_scan<<<dim3((ns + 255) / 256), dim3(256), 0, stream>>>(f_xc, f_xdbl, f_dt, f_xz,
                                                             p[6], p[7], b_y, batch, L);
    cvt(p[8], b_wout, 256, 512, 512, 512);
    gemm(b_y, b_wout, nullptr, outbuf, M, 256, 512, 0);
  };

  const float* x_in = (const float*)d_in[0];
  float* out_f = (float*)d_out;
  size_t S0 = (size_t)Bb * NV * SEG0 * DMD;
  (void)hipMemcpyAsync(out_f, x_in, S0 * 4, hipMemcpyDeviceToDevice, stream); // outs[0] = x
  (void)hipMemcpyAsync(f_x, x_in, S0 * 4, hipMemcpyDeviceToDevice, stream);

  int sn = SEG0;
  size_t out_off = S0;
  for (int layer = 0; layer < 3; ++layer) {
    // insertion-order flattening: x; per layer time(9), dim(9), merge(4 for layer>0)
    int basei = 1 + layer * 18 + ((layer > 1) ? 4 * (layer - 1) : 0);
    const float* tp[9]; const float* dp[9];
    for (int j = 0; j < 9; ++j) {
      tp[j] = (const float*)d_in[basei + j];
      dp[j] = (const float*)d_in[basei + 9 + j];
    }
    if (layer > 0) {
      const float* mg    = (const float*)d_in[basei + 18];
      const float* mb    = (const float*)d_in[basei + 19];
      const float* mw    = (const float*)d_in[basei + 20];
      const float* mbias = (const float*)d_in[basei + 21];
      int rows = Bb * NV * (sn / 2);
      k_merge_norm<<<dim3(rows), dim3(128), 0, stream>>>(f_x, mg, mb, b_xc, rows);
      cvt(mw, b_wlin, 256, 512, 512, 512);
      gemm(b_xc, b_wlin, mbias, f_x, rows, 256, 512, 2);  // bias only
      sn /= 2;
    }
    int M = Bb * NV * sn;
    int tot = M * 256;
    // time stage: rows already [B,D,sn,dm]
    cvt(f_x, b_x, M, 256, 256, 256);
    mamba(tp, b_x, Bb * NV, sn, f_x);
    // dim stage: transpose (D,sn)
    k_tr2b<<<dim3((tot + 255) / 256), dim3(256), 0, stream>>>(f_x, b_x, Bb, NV, sn, 256);
    mamba(dp, b_x, Bb * sn, NV, f_o);
    k_trback<<<dim3((tot + 255) / 256), dim3(256), 0, stream>>>(f_o, f_x, Bb, NV, sn, 256);
    (void)hipMemcpyAsync(out_f + out_off, f_x, (size_t)M * 256 * 4, hipMemcpyDeviceToDevice, stream);
    out_off += (size_t)M * 256;
  }
}
